// ManifoldHyperConnections_50096498541288
// MI455X (gfx1250) — matrix operand order repaired and checked by reading, not compiled, not run on hardware
//
#include <hip/hip_runtime.h>
#include <hip/hip_bf16.h>
#include <math.h>

// ---------------------------------------------------------------------------
// ManifoldHyperConnections for MI455X (gfx1250, wave32, WMMA)
//
// B=4, T=2048, N=4 streams, C=1024  -> bt = 8192 tokens, in_dim = 4096
// Memory-bound (~320 MB moved -> ~14 us @ 23.3 TB/s). 16 tokens per WG,
// stream tile staged once in LDS (f32, 256 KB of the 320 KB WGP LDS),
// projection GEMM (16 x 4096)@(4096 x 32) via v_wmma_f32_16x16x32_bf16 with
// K split across the 8 waves; recombination reads the LDS tile (no second
// HBM pass over stream). Weights pre-packed to bf16 B-fragments by a prep
// kernel (256 KB, L2-resident).
// ---------------------------------------------------------------------------

typedef __attribute__((ext_vector_type(16))) __bf16 v16bf;
typedef __attribute__((ext_vector_type(8)))  float  v8f;

#define BT        8192
#define INDIM     4096
#define C_EMB     1024
#define TILE_T    16          // tokens per workgroup
#define ROWF      4100        // padded LDS row (floats) -> breaks bank collisions
#define NKB       128         // 4096 / 32 K-blocks
#define NFEAT     24          // 4 pre + 4 post + 16 res (padded to 32 in GEMM)

// LDS layout (floats)
#define OFF_TILE   0                        // 16 * 4100           = 65600
#define OFF_RED    (OFF_TILE + TILE_T*ROWF) // 8 waves*2 tiles*256 = 4096
#define OFF_PART   (OFF_RED + 8*2*256)      // 256
#define OFF_RMS    (OFF_PART + 256)         // 16
#define OFF_SFEAT  (OFF_RMS + 16)           // 16*24 = 384
#define OFF_COEF   (OFF_SFEAT + TILE_T*NFEAT) // 16*24 = 384
#define SMEM_FLOATS (OFF_COEF + TILE_T*NFEAT)
#define SMEM_BYTES  (SMEM_FLOATS * sizeof(float))

static __device__ __forceinline__ float4 f4_fma4(float a0, float4 v0, float a1, float4 v1,
                                                 float a2, float4 v2, float a3, float4 v3) {
    float4 r;
    r.x = a0*v0.x + a1*v1.x + a2*v2.x + a3*v3.x;
    r.y = a0*v0.y + a1*v1.y + a2*v2.y + a3*v3.y;
    r.z = a0*v0.z + a1*v1.z + a2*v2.z + a3*v3.z;
    r.w = a0*v0.w + a1*v1.w + a2*v2.w + a3*v3.w;
    return r;
}

static __device__ __forceinline__ float4 f4_axpy(float a, float4 v, float4 b) {
    b.x += a*v.x; b.y += a*v.y; b.z += a*v.z; b.w += a*v.w;
    return b;
}

// ---------------------------------------------------------------------------
// Prep kernel: pack W = [w_pre(4); w_post(4); w_res(16); zeros(8)]  (32 x 4096)
// into bf16 B-fragments for v_wmma_f32_16x16x32_bf16.
// B (32x16, KxN) wave32 layout: lane half hl holds K = 8*hl + [0..7] (elems 0-7)
// and K = 16 + 8*hl + [0..7] (elems 8-15).
// frag index: ((h*128 + kb)*32 + lane)*16 + (v*2 + s)   h = N-half (0/1)
// ---------------------------------------------------------------------------
__global__ __launch_bounds__(256) void mhc_pack_w(
        const float* __restrict__ wpre, const float* __restrict__ wpost,
        const float* __restrict__ wres, __bf16* __restrict__ frag) {
    int gid = blockIdx.x * 256 + threadIdx.x;   // 0 .. 131071
    int idx    = gid & 15;
    int lane   = (gid >> 4) & 31;
    int kb     = (gid >> 9) & 127;
    int h      = (gid >> 16) & 1;
    int N      = lane & 15;
    int half_l = lane >> 4;
    int v      = idx >> 1;
    int s      = idx & 1;
    int Klocal = (v >> 2) * 16 + half_l * 8 + (v & 3) * 2 + s;
    int K      = kb * 32 + Klocal;
    int j      = h * 16 + N;                    // feature row 0..31
    float val = 0.0f;
    if (j < 4)       val = wpre [(size_t)j      * INDIM + K];
    else if (j < 8)  val = wpost[(size_t)(j-4)  * INDIM + K];
    else if (j < 24) val = wres [(size_t)(j-8)  * INDIM + K];
    frag[gid] = (__bf16)val;                    // native v_cvt f32->bf16 (RNE)
}

// ---------------------------------------------------------------------------
// Main kernel: one workgroup = 16 tokens, 256 threads = 8 wave32
// ---------------------------------------------------------------------------
__global__ __launch_bounds__(256) void mhc_main(
        const float* __restrict__ streamp,   // (bt, 4, 1024) = (bt, 4096)
        const float* __restrict__ outp,      // (bt, 1024)
        const __bf16* __restrict__ wfrag,    // packed weight fragments (256 KB)
        const float* __restrict__ bpre,      // (4)
        const float* __restrict__ bpost,     // (4)
        const float* __restrict__ bres,      // (16)
        const float* __restrict__ apre_p,
        const float* __restrict__ apost_p,
        const float* __restrict__ ares_p,
        float* __restrict__ d_out) {         // [x_in (bt*1024) | updated (bt*4096)]

    extern __shared__ float smem[];
    float* tile   = smem + OFF_TILE;    // [16][4100]  raw stream tile (f32)
    float* redbuf = smem + OFF_RED;     // [8 waves][2 tiles][16][16]
    float* part   = smem + OFF_PART;    // [256] sum-of-squares partials
    float* rmsv   = smem + OFF_RMS;     // [16]
    float* sfeat  = smem + OFF_SFEAT;   // [16][24] projected features
    float* coef   = smem + OFF_COEF;    // [16][24] hpre(4) hpost(4) hres(16)

    const int tid  = threadIdx.x;
    const int wv   = tid >> 5;
    const int lane = tid & 31;
    const size_t tileBase = (size_t)blockIdx.x * TILE_T;

    // Prefetch the out tile (needed in Phase 4): 64 KB = 256 cachelines.
    __builtin_prefetch(outp + tileBase * C_EMB + (size_t)tid * 64, 0, 0);

    // ---------------- Phase 1: stage stream tile, compute rms ---------------
    {
        const int token  = tid >> 4;        // 16 threads per token
        const int lane16 = tid & 15;
        const float4* src = (const float4*)(streamp + (tileBase + token) * (size_t)INDIM);
        float* dstrow = &tile[token * ROWF];
        float ss = 0.0f;
        #pragma unroll 4
        for (int i = 0; i < 64; ++i) {
            int q = i * 16 + lane16;        // float4 index within the row
            float4 v = src[q];
            *(float4*)&dstrow[q * 4] = v;
            ss += v.x*v.x + v.y*v.y + v.z*v.z + v.w*v.w;
        }
        part[tid] = ss;
    }
    __syncthreads();
    if (tid < TILE_T) {
        float s = 0.0f;
        #pragma unroll
        for (int k = 0; k < 16; ++k) s += part[tid * 16 + k];
        rmsv[tid] = sqrtf(s * (1.0f / (float)INDIM) + 1e-6f);
    }
    __syncthreads();

    // ---------------- Phase 2: WMMA GEMM  S = tile @ W^T  -------------------
    // CDNA5 16-bit A layout (16x32): lane = M + 16*hl; the lane's 16 K-values
    // are two contiguous 8-float runs: K = kb*32 + 8*hl + [0..7] and
    // K = kb*32 + 16 + 8*hl + [0..7]  -> 4x ds_load_b128 + packed cvt.
    {
        const int Ma = lane & 15;
        const int hl = lane >> 4;
        const float* arow = &tile[Ma * ROWF + 8 * hl];
        const __bf16* bptr0 = &wfrag[((size_t)(0 * NKB) * 32 + lane) * 16];
        const __bf16* bptr1 = &wfrag[((size_t)(1 * NKB) * 32 + lane) * 16];
        v8f acc0 = {};
        v8f acc1 = {};
        for (int i = 0; i < 16; ++i) {
            const int kb = wv * 16 + i;     // this wave's K-blocks
            const float* base = arow + kb * 32;
            float4 c0 = *(const float4*)&base[0];
            float4 c1 = *(const float4*)&base[4];
            float4 c2 = *(const float4*)&base[16];
            float4 c3 = *(const float4*)&base[20];
            v16bf a;
            a[0]  = (__bf16)c0.x; a[1]  = (__bf16)c0.y;
            a[2]  = (__bf16)c0.z; a[3]  = (__bf16)c0.w;
            a[4]  = (__bf16)c1.x; a[5]  = (__bf16)c1.y;
            a[6]  = (__bf16)c1.z; a[7]  = (__bf16)c1.w;
            a[8]  = (__bf16)c2.x; a[9]  = (__bf16)c2.y;
            a[10] = (__bf16)c2.z; a[11] = (__bf16)c2.w;
            a[12] = (__bf16)c3.x; a[13] = (__bf16)c3.y;
            a[14] = (__bf16)c3.z; a[15] = (__bf16)c3.w;

            const v16bf b0 = *(const v16bf*)&bptr0[(size_t)kb * 32 * 16];
            const v16bf b1 = *(const v16bf*)&bptr1[(size_t)kb * 32 * 16];
            acc0 = __builtin_amdgcn_wmma_f32_16x16x32_bf16(false, a, false, b0,
                                                           (short)0, acc0, false, false);
            acc1 = __builtin_amdgcn_wmma_f32_16x16x32_bf16(false, a, false, b1,
                                                           (short)0, acc1, false, false);
        }
        // C layout: VGPR r, lane l -> (M = r + 8*(l>>4), N = l&15)
        #pragma unroll
        for (int r = 0; r < 8; ++r) {
            const int Mr = r + 8 * hl;
            redbuf[((wv * 2 + 0) * 16 + Mr) * 16 + Ma] = acc0[r];
            redbuf[((wv * 2 + 1) * 16 + Mr) * 16 + Ma] = acc1[r];
        }
    }
    __syncthreads();

    // ---------------- cross-wave reduction + alpha/rms/bias -----------------
    {
        const float a_pre  = apre_p[0];
        const float a_post = apost_p[0];
        const float a_res  = ares_p[0];
        for (int idx = tid; idx < 512; idx += 256) {
            const int h = idx >> 8;
            const int rem = idx & 255;
            const int M = rem >> 4;
            const int N = rem & 15;
            const int j = h * 16 + N;
            if (j < NFEAT) {
                float s = 0.0f;
                #pragma unroll
                for (int w2 = 0; w2 < 8; ++w2)
                    s += redbuf[((w2 * 2 + h) * 16 + M) * 16 + N];
                float alpha, bias;
                if (j < 4)       { alpha = a_pre;  bias = bpre[j];      }
                else if (j < 8)  { alpha = a_post; bias = bpost[j - 4]; }
                else             { alpha = a_res;  bias = bres[j - 8];  }
                sfeat[M * NFEAT + j] = alpha * s / rmsv[M] + bias;
            }
        }
    }
    __syncthreads();

    // ---------------- Phase 3: sigmoid + sinkhorn (1 thread / token) --------
    if (tid < TILE_T) {
        const float* f = &sfeat[tid * NFEAT];
        float* cf = &coef[tid * NFEAT];
        #pragma unroll
        for (int n = 0; n < 4; ++n) {
            cf[n]     = 1.0f / (1.0f + expf(-f[n]));          // h_pre
            cf[4 + n] = 2.0f / (1.0f + expf(-f[4 + n]));      // h_post
        }
        float m[16];
        float mx = f[8];
        #pragma unroll
        for (int k = 1; k < 16; ++k) mx = fmaxf(mx, f[8 + k]);
        #pragma unroll
        for (int k = 0; k < 16; ++k) m[k] = expf(f[8 + k] - mx);
        #pragma unroll
        for (int it = 0; it < 5; ++it) {
            #pragma unroll
            for (int r = 0; r < 4; ++r) {
                float s = m[r*4] + m[r*4+1] + m[r*4+2] + m[r*4+3] + 1e-6f;
                float inv = 1.0f / s;
                m[r*4] *= inv; m[r*4+1] *= inv; m[r*4+2] *= inv; m[r*4+3] *= inv;
            }
            #pragma unroll
            for (int c = 0; c < 4; ++c) {
                float s = m[c] + m[4+c] + m[8+c] + m[12+c] + 1e-6f;
                float inv = 1.0f / s;
                m[c] *= inv; m[4+c] *= inv; m[8+c] *= inv; m[12+c] *= inv;
            }
        }
        #pragma unroll
        for (int k = 0; k < 16; ++k) cf[8 + k] = m[k];        // h_res[n*4+m]
    }
    __syncthreads();

    // ---------------- Phase 4: recombine streams (2 tokens / wave) ----------
    {
        float* xin_out = d_out;
        float* upd_out = d_out + (size_t)BT * C_EMB;
        for (int tt = 0; tt < 2; ++tt) {
            const int t = wv * 2 + tt;
            const float* cf = &coef[t * NFEAT];
            float hp[4], hq[4], hr[16];
            #pragma unroll
            for (int n = 0; n < 4; ++n) { hp[n] = cf[n]; hq[n] = cf[4 + n]; }
            #pragma unroll
            for (int k = 0; k < 16; ++k) hr[k] = cf[8 + k];

            const size_t tg = tileBase + t;
            const float*  row = &tile[t * ROWF];
            const float4* o4  = (const float4*)(outp + tg * C_EMB);
            float4* xo = (float4*)(xin_out + tg * C_EMB);
            float4* uo = (float4*)(upd_out + tg * (size_t)INDIM);

            #pragma unroll 2
            for (int i = 0; i < 8; ++i) {
                const int c4 = lane + 32 * i;          // float4 index 0..255
                float4 s0 = *(const float4*)&row[0 * C_EMB + 4 * c4];
                float4 s1 = *(const float4*)&row[1 * C_EMB + 4 * c4];
                float4 s2 = *(const float4*)&row[2 * C_EMB + 4 * c4];
                float4 s3 = *(const float4*)&row[3 * C_EMB + 4 * c4];
                float4 ov = o4[c4];

                xo[c4] = f4_fma4(hp[0], s0, hp[1], s1, hp[2], s2, hp[3], s3);
                #pragma unroll
                for (int n = 0; n < 4; ++n) {
                    float4 u = f4_fma4(hr[n*4+0], s0, hr[n*4+1], s1,
                                       hr[n*4+2], s2, hr[n*4+3], s3);
                    uo[n * 256 + c4] = f4_axpy(hq[n], ov, u);
                }
            }
        }
    }
}

// ---------------------------------------------------------------------------
extern "C" void kernel_launch(void* const* d_in, const int* in_sizes, int n_in,
                              void* d_out, int out_size, void* d_ws, size_t ws_size,
                              hipStream_t stream) {
    const float* streamp = (const float*)d_in[0];
    const float* outp    = (const float*)d_in[1];
    const float* w_pre   = (const float*)d_in[2];
    const float* w_post  = (const float*)d_in[3];
    const float* w_res   = (const float*)d_in[4];
    const float* b_pre   = (const float*)d_in[5];
    const float* b_post  = (const float*)d_in[6];
    const float* b_res   = (const float*)d_in[7];
    const float* a_pre   = (const float*)d_in[8];
    const float* a_post  = (const float*)d_in[9];
    const float* a_res   = (const float*)d_in[10];

    __bf16* wfrag = (__bf16*)d_ws;      // 2*128*32*16 bf16 = 256 KB

    // pack weights into WMMA B fragments (131072 elements)
    mhc_pack_w<<<512, 256, 0, stream>>>(w_pre, w_post, w_res, wfrag);

    // main fused kernel: 8192 tokens / 16 per WG
    mhc_main<<<BT / TILE_T, 256, SMEM_BYTES, stream>>>(
        streamp, outp, wfrag, b_pre, b_post, b_res,
        a_pre, a_post, a_res, (float*)d_out);
}
